// ResGCN_5772436045963
// MI455X (gfx1250) — matrix operand order repaired and checked by reading, not compiled, not run on hardware
//
#include <hip/hip_runtime.h>
#include <hip/hip_bf16.h>
#include <math.h>

// ---------------------------------------------------------------------------
// ResGCN for MI455X (gfx1250, wave32, WMMA), atomic-free aggregation.
//
//  once per call:  CSC build (hist -> scan -> fill -> per-bucket sort)
//  per layer:      pack_w : W -> WMMA B-fragment layout (bf16 hi/lo, in L2)
//                  gemm   : S = H @ W      (3x v_wmma bf16 split, f32 accum)
//                  gather : H = relu(sum_in(ew * S[src]) + b) [+ residual]
//  final layer:    gather40 + fused log_softmax -> d_out
// ---------------------------------------------------------------------------

typedef __attribute__((ext_vector_type(16))) __bf16 v16bf;
typedef __attribute__((ext_vector_type(8)))  float  v8f;

#define KDIM 128
#define ROWS_PER_BLOCK 128

__device__ inline void splitf(float f, __bf16& hi, __bf16& lo) {
    __bf16 h = (__bf16)f;
    hi = h;
    lo = (__bf16)(f - (float)h);
}

// ---------------------------------------------------------------------------
// pack_w: swizzle W[128 x nOut] into per-lane WMMA B fragments (hi/lo bf16).
// Fragment f = nt*4 + kc, lane l: 16 bf16 = B[K = kc*32 + 16*(l>>4) + e][nt*16 + (l&15)]
// Stored contiguously: P[(f*32 + l)*16 + e]. One small launch per layer; lives in L2.
// ---------------------------------------------------------------------------
__global__ __launch_bounds__(256) void pack_w(const float* __restrict__ W,
                                              __bf16* __restrict__ PH,
                                              __bf16* __restrict__ PL, int nOut) {
    const int ntiles = (nOut + 15) >> 4;
    const int total  = ntiles * 4 * 32;          // fragment-lanes
    for (int i = threadIdx.x; i < total; i += 256) {
        const int lane = i & 31;
        const int kc   = (i >> 5) & 3;
        const int nt   = i >> 7;
        const int hf   = lane >> 4;
        const int n    = nt * 16 + (lane & 15);
        v16bf hv, lv;
#pragma unroll
        for (int e = 0; e < 16; ++e) {
            const int K = kc * 32 + 16 * hf + e;
            float w = (n < nOut) ? W[K * nOut + n] : 0.0f;
            __bf16 h, l;
            splitf(w, h, l);
            hv[e] = h; lv[e] = l;
        }
        ((v16bf*)PH)[i] = hv;
        ((v16bf*)PL)[i] = lv;
    }
}

// ---------------------------------------------------------------------------
// GEMM: S[nRows x nOut] = A[nRows x 128] * W (via packed fragments).
// 256 threads = 8 waves; each wave computes 16 rows x nOut cols.
// No LDS: B fragments stream from L2 as b128 loads.
// ---------------------------------------------------------------------------
__global__ __launch_bounds__(256) void gcn_gemm(const float* __restrict__ A,
                                                const __bf16* __restrict__ PH,
                                                const __bf16* __restrict__ PL,
                                                float* __restrict__ S,
                                                int nRows, int nOut) {
    const int tid    = threadIdx.x;
    const int ntiles = (nOut + 15) >> 4;
    const int wave   = tid >> 5;
    const int lane   = tid & 31;
    const int hf     = lane >> 4;
    const int m      = lane & 15;
    const int rowBase = blockIdx.x * ROWS_PER_BLOCK + wave * 16;
    if (rowBase >= nRows) return;            // wave-uniform exit
    const bool fullTile = (rowBase + 16 <= nRows);   // wave-uniform

    // A fragments: element e holds k = kc*32 + 16*hf + e (contiguous, same order as pack_w)
    v16bf ahi[4], alo[4];
    {
        int row = rowBase + m;
        if (row >= nRows) row = nRows - 1;
        const float* ar = A + (size_t)row * KDIM;
#pragma unroll
        for (int kc = 0; kc < 4; ++kc) {
            const int k0 = kc * 32 + hf * 16;
#pragma unroll
            for (int e = 0; e < 8; ++e) {
                __bf16 h, l;
                splitf(ar[k0 + e], h, l);
                ahi[kc][e] = h; alo[kc][e] = l;
                splitf(ar[k0 + 8 + e], h, l);
                ahi[kc][e + 8] = h; alo[kc][e + 8] = l;
            }
        }
    }

    const v16bf* fh = (const v16bf*)PH;
    const v16bf* fl = (const v16bf*)PL;

    for (int nt = 0; nt < ntiles; ++nt) {
        const int col = nt * 16 + m;
        v8f acc = {};
#pragma unroll
        for (int kc = 0; kc < 4; ++kc) {
            const int f = (nt * 4 + kc) * 32 + lane;
            const v16bf bh = fh[f];
            const v16bf bl = fl[f];
            acc = __builtin_amdgcn_wmma_f32_16x16x32_bf16(false, ahi[kc], false, bh,
                                                          (short)0, acc, false, false);
            acc = __builtin_amdgcn_wmma_f32_16x16x32_bf16(false, ahi[kc], false, bl,
                                                          (short)0, acc, false, false);
            acc = __builtin_amdgcn_wmma_f32_16x16x32_bf16(false, alo[kc], false, bh,
                                                          (short)0, acc, false, false);
        }
        if (col < nOut) {
            // C/D layout: VGPR r -> M = r + 8*hf
            float* sp = S + (size_t)(rowBase + hf * 8) * nOut + col;
            if (fullTile) {
#pragma unroll
                for (int r = 0; r < 8; ++r) sp[(size_t)r * nOut] = acc[r];
            } else {
#pragma unroll
                for (int r = 0; r < 8; ++r)
                    if (rowBase + hf * 8 + r < nRows) sp[(size_t)r * nOut] = acc[r];
            }
        }
    }
}

// ---------------------------------------------------------------------------
// CSC build: counts/rank histogram, prefix scan, fill, deterministic sort.
// Edge records packed as int2 {src, bitcast(weight)} for single-b64 fetch.
// ---------------------------------------------------------------------------
__global__ __launch_bounds__(256) void hist_kernel(const int* __restrict__ dst,
                                                   int* __restrict__ counts,
                                                   int* __restrict__ rank, int nEdges) {
    int e = blockIdx.x * 256 + threadIdx.x;
    if (e < nEdges) rank[e] = atomicAdd(&counts[dst[e]], 1);
}

__global__ __launch_bounds__(1024) void scan_kernel(const int* __restrict__ counts,
                                                    int* __restrict__ rowptr, int nNodes) {
    __shared__ int buf[1024];
    __shared__ int carry;
    if (threadIdx.x == 0) carry = 0;
    __syncthreads();
    for (int base = 0; base < nNodes; base += 1024) {
        const int i = base + threadIdx.x;
        int v = (i < nNodes) ? counts[i] : 0;
        buf[threadIdx.x] = v;
        __syncthreads();
#pragma unroll
        for (int off = 1; off < 1024; off <<= 1) {
            int t = (threadIdx.x >= off) ? buf[threadIdx.x - off] : 0;
            __syncthreads();
            buf[threadIdx.x] += t;
            __syncthreads();
        }
        if (i < nNodes) rowptr[i + 1] = carry + buf[threadIdx.x];
        __syncthreads();
        if (threadIdx.x == 1023) carry += buf[1023];
        __syncthreads();
    }
    if (threadIdx.x == 0) rowptr[0] = 0;
}

__global__ __launch_bounds__(256) void fill_kernel(const int* __restrict__ src,
                                                   const int* __restrict__ dst,
                                                   const float* __restrict__ ew,
                                                   const int* __restrict__ rowptr,
                                                   const int* __restrict__ rank,
                                                   int* __restrict__ eidS,
                                                   int2* __restrict__ erec, int nEdges) {
    int e = blockIdx.x * 256 + threadIdx.x;
    if (e >= nEdges) return;
    int pos = rowptr[dst[e]] + rank[e];
    eidS[pos] = e;
    int2 r;
    r.x = src[e];
    r.y = __float_as_int(ew[e]);
    erec[pos] = r;
}

// selection-sort each node's bucket by original edge id -> bitwise-deterministic sums
__global__ __launch_bounds__(256) void sort_buckets(int* __restrict__ eidS,
                                                    int2* __restrict__ erec,
                                                    const int* __restrict__ rowptr,
                                                    int nNodes) {
    int n = blockIdx.x * 256 + threadIdx.x;
    if (n >= nNodes) return;
    const int a = rowptr[n], b = rowptr[n + 1];
    for (int i = a; i < b - 1; ++i) {
        int mi = i, mv = eidS[i];
        for (int j = i + 1; j < b; ++j) {
            int v = eidS[j];
            if (v < mv) { mv = v; mi = j; }
        }
        if (mi != i) {
            int te = eidS[i]; eidS[i] = eidS[mi]; eidS[mi] = te;
            int2 tr = erec[i]; erec[i] = erec[mi]; erec[mi] = tr;
        }
    }
}

// ---------------------------------------------------------------------------
// Gather-aggregate (dim 128): one wave per node, 4 cols/lane, zero atomics.
//   out = relu(sum_j ew*S[src] + bias) [+ res]
// ---------------------------------------------------------------------------
__global__ __launch_bounds__(256) void gcn_gather(const float* __restrict__ S,
                                                  const int* __restrict__ rowptr,
                                                  const int2* __restrict__ erec,
                                                  const float* __restrict__ bias,
                                                  const float* __restrict__ res,
                                                  float* __restrict__ out, int nNodes) {
    const int node = (blockIdx.x * 256 + threadIdx.x) >> 5;
    if (node >= nNodes) return;
    const int lane = threadIdx.x & 31;
    const int c = lane * 4;
    const int a = rowptr[node], b = rowptr[node + 1];
    float a0 = 0.f, a1 = 0.f, a2 = 0.f, a3 = 0.f;
    for (int j = a; j < b; ++j) {
        const int2  r = erec[j];
        const float w = __int_as_float(r.y);
        const float4 v = *(const float4*)(S + (size_t)r.x * KDIM + c);
        a0 += w * v.x; a1 += w * v.y; a2 += w * v.z; a3 += w * v.w;
    }
    const float4 bv = *(const float4*)(bias + c);
    a0 += bv.x; a1 += bv.y; a2 += bv.z; a3 += bv.w;
    a0 = a0 > 0.f ? a0 : 0.f;
    a1 = a1 > 0.f ? a1 : 0.f;
    a2 = a2 > 0.f ? a2 : 0.f;
    a3 = a3 > 0.f ? a3 : 0.f;
    if (res) {
        const float4 rv = *(const float4*)(res + (size_t)node * KDIM + c);
        a0 += rv.x; a1 += rv.y; a2 += rv.z; a3 += rv.w;
    }
    float4 o; o.x = a0; o.y = a1; o.z = a2; o.w = a3;
    *(float4*)(out + (size_t)node * KDIM + c) = o;
}

// ---------------------------------------------------------------------------
// Final layer: gather (dim ncls=40) + bias + relu + log_softmax, fused.
// One wave per node; lane covers cols {lane, lane+32}.
// ---------------------------------------------------------------------------
__global__ __launch_bounds__(256) void gcn_final(const float* __restrict__ S,
                                                 const int* __restrict__ rowptr,
                                                 const int2* __restrict__ erec,
                                                 const float* __restrict__ b,
                                                 float* __restrict__ out,
                                                 int nNodes, int ncls) {
    const int node = (blockIdx.x * 256 + threadIdx.x) >> 5;
    if (node >= nNodes) return;
    const int lane = threadIdx.x & 31;
    const int c1 = lane + 32;
    const int ea = rowptr[node], eb = rowptr[node + 1];
    float acc0 = 0.f, acc1 = 0.f;
    for (int j = ea; j < eb; ++j) {
        const int2  r = erec[j];
        const float w = __int_as_float(r.y);
        const float* row = S + (size_t)r.x * ncls;
        if (lane < ncls) acc0 += w * row[lane];
        if (c1 < ncls)   acc1 += w * row[c1];
    }
    float v0 = -INFINITY, v1 = -INFINITY;
    if (lane < ncls) { v0 = acc0 + b[lane]; v0 = v0 > 0.f ? v0 : 0.f; }
    if (c1 < ncls)   { v1 = acc1 + b[c1];   v1 = v1 > 0.f ? v1 : 0.f; }

    float mx = fmaxf(v0, v1);
#pragma unroll
    for (int off = 16; off > 0; off >>= 1) mx = fmaxf(mx, __shfl_xor(mx, off, 32));
    float s = 0.f;
    if (lane < ncls) s += expf(v0 - mx);
    if (c1 < ncls)   s += expf(v1 - mx);
#pragma unroll
    for (int off = 16; off > 0; off >>= 1) s += __shfl_xor(s, off, 32);
    const float lse = logf(s) + mx;

    if (lane < ncls) out[(size_t)node * ncls + lane] = v0 - lse;
    if (c1 < ncls)   out[(size_t)node * ncls + c1]   = v1 - lse;
}

// ---------------------------------------------------------------------------
// Inputs: 0:x 1:src 2:dst 3:edge_weight 4:W1 5:W_hid 6:W10 7:b1 8:b_hid 9:b10
// ---------------------------------------------------------------------------
extern "C" void kernel_launch(void* const* d_in, const int* in_sizes, int n_in,
                              void* d_out, int out_size, void* d_ws, size_t ws_size,
                              hipStream_t stream) {
    const float* x    = (const float*)d_in[0];
    const int*   src  = (const int*)d_in[1];
    const int*   dst  = (const int*)d_in[2];
    const float* ew   = (const float*)d_in[3];
    const float* W1   = (const float*)d_in[4];
    const float* Whid = (const float*)d_in[5];
    const float* W10  = (const float*)d_in[6];
    const float* b1   = (const float*)d_in[7];
    const float* bhid = (const float*)d_in[8];
    const float* b10  = (const float*)d_in[9];

    const int nNodes = in_sizes[0] / KDIM;
    const int nEdges = in_sizes[1];
    const int ncls   = in_sizes[9];          // 40

    // ---- workspace carve-up (keep 32B alignment for v16bf fragments) ----
    char* ws = (char*)d_ws;
    float* H  = (float*)ws;                 ws += (size_t)nNodes * KDIM * sizeof(float);
    float* S  = (float*)ws;                 ws += (size_t)nNodes * KDIM * sizeof(float);
    __bf16* PH = (__bf16*)ws;               ws += 16384 * sizeof(__bf16);
    __bf16* PL = (__bf16*)ws;               ws += 16384 * sizeof(__bf16);
    int2* erec  = (int2*)ws;                ws += (size_t)nEdges * sizeof(int2);
    int* counts = (int*)ws;                 ws += (size_t)nNodes * sizeof(int);
    int* rowptr = (int*)ws;                 ws += ((size_t)nNodes + 2) * sizeof(int);
    int* rank   = (int*)ws;                 ws += (size_t)nEdges * sizeof(int);
    int* eidS   = (int*)ws;                 ws += (size_t)nEdges * sizeof(int);

    const dim3 blk(256);
    const int gemmGrid = (nNodes + ROWS_PER_BLOCK - 1) / ROWS_PER_BLOCK;
    const int edgeGrid = (nEdges + 255) / 256;
    const int nodeGrid = (nNodes + 255) / 256;
    const int waveGrid = (nNodes + 7) / 8;   // 1 wave per node, 8 waves/block

    // ---- CSC build (deterministic after per-bucket sort) ----
    hipMemsetAsync(counts, 0, (size_t)nNodes * sizeof(int), stream);
    hist_kernel<<<edgeGrid, blk, 0, stream>>>(dst, counts, rank, nEdges);
    scan_kernel<<<1, 1024, 0, stream>>>(counts, rowptr, nNodes);
    fill_kernel<<<edgeGrid, blk, 0, stream>>>(src, dst, ew, rowptr, rank,
                                              eidS, erec, nEdges);
    sort_buckets<<<nodeGrid, blk, 0, stream>>>(eidS, erec, rowptr, nNodes);

    // ---- layer 1: x -> H ----
    pack_w<<<1, blk, 0, stream>>>(W1, PH, PL, KDIM);
    gcn_gemm<<<gemmGrid, blk, 0, stream>>>(x, PH, PL, S, nNodes, KDIM);
    gcn_gather<<<waveGrid, blk, 0, stream>>>(S, rowptr, erec, b1, nullptr, H, nNodes);

    // ---- layers 2..9: residual hidden layers ----
    for (int i = 0; i < 8; ++i) {
        pack_w<<<1, blk, 0, stream>>>(Whid + (size_t)i * KDIM * KDIM, PH, PL, KDIM);
        gcn_gemm<<<gemmGrid, blk, 0, stream>>>(H, PH, PL, S, nNodes, KDIM);
        gcn_gather<<<waveGrid, blk, 0, stream>>>(S, rowptr, erec,
                                                 bhid + (size_t)i * KDIM, H, H, nNodes);
    }

    // ---- layer 10: logits + log_softmax ----
    pack_w<<<1, blk, 0, stream>>>(W10, PH, PL, ncls);
    gcn_gemm<<<gemmGrid, blk, 0, stream>>>(H, PH, PL, S, nNodes, ncls);
    gcn_final<<<waveGrid, blk, 0, stream>>>(S, rowptr, erec, b10,
                                            (float*)d_out, nNodes, ncls);
}
